// SimilarityEmbedding_30434138259836
// MI455X (gfx1250) — matrix operand order repaired and checked by reading, not compiled, not run on hardware
//
#include <hip/hip_runtime.h>

// ---------------------------------------------------------------------------
// SimilarityEmbedding loss, MI455X (gfx1250, wave32, WMMA)
//
// loss = sum_{i,j} [ same(i,j) ? 1 - r_ij : max(r_ij, 0) ]
//   r_ij = (S[i].T[j]) * denom_inv[i],  denom_inv[i] = 1/(|S[i]|^2 * |T[i]|^2)
//
// GEMM done as bf16x3 split-precision WMMA (hi/lo decomposition of fp32):
//   S.T' ~= Shi.Thi' + Shi.Tlo' + Slo.Thi'   via v_wmma_f32_16x16x32_bf16
//
// Workspace layout (needs ~16.05 MiB):
//   [0,  4Mi)  Shi  bf16[8192][256]
//   [4Mi,8Mi)  Slo
//   [8Mi,12Mi) Thi
//   [12Mi,16Mi)Tlo
//   [16Mi, +32Ki)        denom_inv f32[8192]
//   [16Mi+32Ki, +16Ki)   block partial sums f32[4096]
// ---------------------------------------------------------------------------

#define NROWS 8192
#define DDIM  256

typedef __bf16 bf16_t;
typedef bf16_t v16bf __attribute__((ext_vector_type(16)));
typedef bf16_t v8bf  __attribute__((ext_vector_type(8)));
typedef float  v8f   __attribute__((ext_vector_type(8)));

static __device__ __forceinline__ v16bf pack16(v8bf a, v8bf b) {
  union { v16bf v; v8bf h[2]; } u;
  u.h[0] = a; u.h[1] = b;
  return u.v;
}

// A fragment: 16x32 bf16 tile starting at (row0, k0) of a row-major [NROWS][DDIM]
// matrix. ISA layout (CDNA5 ISA, 16-bit A 16x32): lane L<16 holds M=L,
// K = k0+{0..7, 16..23}; lane L+16 holds M=L, K = k0+{8..15, 24..31}.
static __device__ __forceinline__ v16bf load_fragA(const bf16_t* M, int row0,
                                                   int k0, int lane) {
  int r = lane & 15, h = lane >> 4;
  const bf16_t* p = M + (size_t)(row0 + r) * DDIM + k0 + h * 8;
  v8bf f0 = *(const v8bf*)p;          // K = k0 + h*8      .. +7
  v8bf f1 = *(const v8bf*)(p + 16);   // K = k0 + 16 + h*8 .. +7
  return pack16(f0, f1);
}

// B fragment: 32x16 bf16 tile, B[k][n] = T[col0+n][k0+k] (T row-major).
// ISA layout (matches A): lane L<16 holds N=L, K = k0+{0..7, 16..23};
// lane L+16 holds N=L, K = k0+{8..15, 24..31}.
static __device__ __forceinline__ v16bf load_fragB(const bf16_t* T, int col0,
                                                   int k0, int lane) {
  int n = lane & 15, h = lane >> 4;
  const bf16_t* p = T + (size_t)(col0 + n) * DDIM + k0 + h * 8;
  v8bf f0 = *(const v8bf*)p;
  v8bf f1 = *(const v8bf*)(p + 16);
  return pack16(f0, f1);
}

static __device__ __forceinline__ v8f wmma_bf16(v16bf a, v16bf b, v8f c) {
  return __builtin_amdgcn_wmma_f32_16x16x32_bf16(
      /*neg_a=*/false, a, /*neg_b=*/false, b,
      /*c_mod=*/(short)0, c, /*reuse_a=*/false, /*reuse_b=*/false);
}

// ---------------------------------------------------------------------------
// Pass 1: bf16 hi/lo split + denom_inv. One 256-thread block per row.
// ---------------------------------------------------------------------------
__global__ __launch_bounds__(256) void prep_kernel(
    const float* __restrict__ src, const float* __restrict__ tgt,
    bf16_t* __restrict__ shi, bf16_t* __restrict__ slo,
    bf16_t* __restrict__ thi, bf16_t* __restrict__ tlo,
    float* __restrict__ denom_inv) {
  int i = blockIdx.x;
  int k = threadIdx.x;
  size_t idx = (size_t)i * DDIM + k;

  float s = src[idx];
  float t = tgt[idx];

  bf16_t sh = (bf16_t)s;
  bf16_t th = (bf16_t)t;
  bf16_t sl = (bf16_t)(s - (float)sh);
  bf16_t tl = (bf16_t)(t - (float)th);
  shi[idx] = sh; slo[idx] = sl;
  thi[idx] = th; tlo[idx] = tl;

  __shared__ float rs[256];
  __shared__ float rt[256];
  rs[k] = s * s;
  rt[k] = t * t;
  __syncthreads();
#pragma unroll
  for (int sft = 128; sft > 0; sft >>= 1) {
    if (k < sft) { rs[k] += rs[k + sft]; rt[k] += rt[k + sft]; }
    __syncthreads();
  }
  if (k == 0) denom_inv[i] = 1.0f / (rs[0] * rt[0]);
}

// ---------------------------------------------------------------------------
// Pass 2: fused GEMM + loss. 64x64 grid of 128x128 tiles, 8 waves/block.
// Wave w covers rows [wm*32, +32) x cols [wn*64, +64) of its block tile
// (wm = w&3, wn = w>>2): 2x4 subtiles of 16x16, bf16x3 -> 24 WMMAs per K=32.
// ---------------------------------------------------------------------------
__global__ __launch_bounds__(256) void gemm_loss_kernel(
    const bf16_t* __restrict__ shi, const bf16_t* __restrict__ slo,
    const bf16_t* __restrict__ thi, const bf16_t* __restrict__ tlo,
    const float* __restrict__ denom_inv,
    const int* __restrict__ label_s, const int* __restrict__ label_t,
    float* __restrict__ partials) {
  int lane = threadIdx.x & 31;
  int wave = threadIdx.x >> 5;
  int wm = wave & 3;
  int wn = wave >> 2;
  int row0 = blockIdx.y * 128 + wm * 32;
  int col0 = blockIdx.x * 128 + wn * 64;

  v8f acc[2][4];
#pragma unroll
  for (int mi = 0; mi < 2; ++mi)
#pragma unroll
    for (int ni = 0; ni < 4; ++ni) acc[mi][ni] = (v8f)0.0f;

#pragma unroll
  for (int k0 = 0; k0 < DDIM; k0 += 32) {
    v16bf ah[2], al[2], bh[4], bl[4];
#pragma unroll
    for (int mi = 0; mi < 2; ++mi) {
      ah[mi] = load_fragA(shi, row0 + 16 * mi, k0, lane);
      al[mi] = load_fragA(slo, row0 + 16 * mi, k0, lane);
    }
#pragma unroll
    for (int ni = 0; ni < 4; ++ni) {
      bh[ni] = load_fragB(thi, col0 + 16 * ni, k0, lane);
      bl[ni] = load_fragB(tlo, col0 + 16 * ni, k0, lane);
    }
#pragma unroll
    for (int mi = 0; mi < 2; ++mi) {
#pragma unroll
      for (int ni = 0; ni < 4; ++ni) {
        acc[mi][ni] = wmma_bf16(ah[mi], bh[ni], acc[mi][ni]);  // hi*hi
        acc[mi][ni] = wmma_bf16(al[mi], bh[ni], acc[mi][ni]);  // lo*hi
        acc[mi][ni] = wmma_bf16(ah[mi], bl[ni], acc[mi][ni]);  // hi*lo
      }
    }
  }

  // Epilogue: C layout (32-bit 16x16): lane<16 -> N=lane, VGPR r -> M=r;
  // lane>=16 -> N=lane-16, VGPR r -> M=8+r.
  int nIdx = lane & 15;
  int rBase = (lane >> 4) * 8;
  float lsum = 0.0f;
#pragma unroll
  for (int mi = 0; mi < 2; ++mi) {
    int lsr[8];
    float dinv[8];
#pragma unroll
    for (int r = 0; r < 8; ++r) {
      int rowg = row0 + mi * 16 + rBase + r;
      lsr[r] = label_s[rowg];
      dinv[r] = denom_inv[rowg];
    }
#pragma unroll
    for (int ni = 0; ni < 4; ++ni) {
      int colg = col0 + ni * 16 + nIdx;
      int ltc = label_t[colg];
#pragma unroll
      for (int r = 0; r < 8; ++r) {
        float v = acc[mi][ni][r] * dinv[r];
        lsum += (lsr[r] == ltc) ? (1.0f - v) : fmaxf(v, 0.0f);
      }
    }
  }

  __shared__ float red[256];
  red[threadIdx.x] = lsum;
  __syncthreads();
#pragma unroll
  for (int sft = 128; sft > 0; sft >>= 1) {
    if (threadIdx.x < sft) red[threadIdx.x] += red[threadIdx.x + sft];
    __syncthreads();
  }
  if (threadIdx.x == 0)
    partials[blockIdx.y * gridDim.x + blockIdx.x] = red[0];
}

// ---------------------------------------------------------------------------
// Pass 3: deterministic fixed-order reduction of 4096 block partials.
// ---------------------------------------------------------------------------
__global__ __launch_bounds__(256) void finalize_kernel(
    const float* __restrict__ partials, float* __restrict__ out) {
  float s = 0.0f;
  for (int i = threadIdx.x; i < 4096; i += 256) s += partials[i];
  __shared__ float red[256];
  red[threadIdx.x] = s;
  __syncthreads();
#pragma unroll
  for (int sft = 128; sft > 0; sft >>= 1) {
    if (threadIdx.x < sft) red[threadIdx.x] += red[threadIdx.x + sft];
    __syncthreads();
  }
  if (threadIdx.x == 0) out[0] = red[0];
}

// ---------------------------------------------------------------------------
extern "C" void kernel_launch(void* const* d_in, const int* in_sizes, int n_in,
                              void* d_out, int out_size, void* d_ws,
                              size_t ws_size, hipStream_t stream) {
  const float* src = (const float*)d_in[0];
  const float* tgt = (const float*)d_in[1];
  const int* label_s = (const int*)d_in[2];
  const int* label_t = (const int*)d_in[3];

  char* ws = (char*)d_ws;
  const size_t MATB = (size_t)NROWS * DDIM * sizeof(bf16_t);  // 4 MiB
  bf16_t* shi = (bf16_t*)(ws + 0 * MATB);
  bf16_t* slo = (bf16_t*)(ws + 1 * MATB);
  bf16_t* thi = (bf16_t*)(ws + 2 * MATB);
  bf16_t* tlo = (bf16_t*)(ws + 3 * MATB);
  float* denom_inv = (float*)(ws + 4 * MATB);
  float* partials = (float*)(ws + 4 * MATB + NROWS * sizeof(float));

  prep_kernel<<<NROWS, 256, 0, stream>>>(src, tgt, shi, slo, thi, tlo,
                                         denom_inv);

  dim3 grid(64, 64);
  gemm_loss_kernel<<<grid, 256, 0, stream>>>(shi, slo, thi, tlo, denom_inv,
                                             label_s, label_t, partials);

  finalize_kernel<<<1, 256, 0, stream>>>(partials, (float*)d_out);
}
